// EnhancedMSTSN_73864847557263
// MI455X (gfx1250) — matrix operand order repaired and checked by reading, not compiled, not run on hardware
//
#include <hip/hip_runtime.h>

typedef __attribute__((ext_vector_type(16))) _Float16 v16h;
typedef __attribute__((ext_vector_type(8)))  _Float16 v8h;
typedef __attribute__((ext_vector_type(8)))  float    v8f;

#define DEV __device__ __forceinline__

DEV v8f wmma16(v16h a, v16h b, v8f c) {
  return __builtin_amdgcn_wmma_f32_16x16x32_f16(false, a, false, b, (short)0, c, false, false);
}
DEV float lrelu(float x) { return x > 0.f ? x : 0.2f * x; }
DEV float gelu_exact(float x) { return 0.5f * x * (1.f + erff(x * 0.70710678118654752f)); }

DEV v16h cat16(v8h lo, v8h hi) {
  return __builtin_shufflevector(lo, hi, 0, 1, 2, 3, 4, 5, 6, 7, 8, 9, 10, 11, 12, 13, 14, 15);
}
// A fragment: 16 contiguous halves at p (K = e + 16*half)
DEV v16h loadA16(const _Float16* p) {
  return cat16(*(const v8h*)p, *(const v8h*)(p + 8));
}
// B fragment from transposed source: 16 contiguous halves at p (K = e + 16*half)
DEV v16h loadBT16(const _Float16* p) {
  return cat16(*(const v8h*)p, *(const v8h*)(p + 8));
}

// ---------------- param f32 -> f16 (optionally transposed) ----------------
__global__ void k_f32_to_f16(const float* __restrict__ in, _Float16* __restrict__ out, int n) {
  int i = blockIdx.x * blockDim.x + threadIdx.x;
  if (i < n) out[i] = (_Float16)in[i];
}
// out[c*R + r] = in[r*C + c]
__global__ void k_f32_to_f16_t(const float* __restrict__ in, _Float16* __restrict__ out,
                               int R, int C) {
  int i = blockIdx.x * blockDim.x + threadIdx.x;
  if (i >= R * C) return;
  int r = i / C, c = i % C;
  out[c * R + r] = (_Float16)in[i];
}

// ---------------- GAT1 prep: h1 = x@W, src/dst scores ----------------
__global__ void k_gat1_prep(const float* __restrict__ x, const float* __restrict__ w,
                            const float* __restrict__ asrc, const float* __restrict__ adst,
                            _Float16* __restrict__ H16, float* __restrict__ src,
                            float* __restrict__ dst) {
  int idx = blockIdx.x * blockDim.x + threadIdx.x;   // g*1024 + n
  if (idx >= 32 * 1024) return;
  const float* xp = x + (size_t)idx * 3;
  float x0 = xp[0], x1 = xp[1], x2 = xp[2];
  float sa[4] = {0.f, 0.f, 0.f, 0.f}, da[4] = {0.f, 0.f, 0.f, 0.f};
  _Float16* hp = H16 + (size_t)idx * 128;
  for (int k = 0; k < 128; ++k) {
    float h = x0 * w[k] + x1 * w[128 + k] + x2 * w[256 + k];
    hp[k] = (_Float16)h;
    int hd = k >> 5;
    sa[hd] += h * asrc[k];
    da[hd] += h * adst[k];
  }
#pragma unroll
  for (int hd = 0; hd < 4; ++hd) {
    src[(size_t)idx * 4 + hd] = sa[hd];
    dst[(size_t)idx * 4 + hd] = da[hd];
  }
}

// ---------------- fused flash GAT aggregation (softmax(rank-1 logits) @ H) ----
#define HST 1032   // padded LDS row stride (halves): mult of 8, rotates banks
template <bool RELU, bool OUT16>
__global__ void k_gat_agg(const _Float16* __restrict__ H, const float* __restrict__ src,
                          const float* __restrict__ dst, const float* __restrict__ bias,
                          float* __restrict__ outf, _Float16* __restrict__ outh, int NH) {
  extern __shared__ char sm[];
  _Float16* HsT = (_Float16*)sm;                      // 32 x HST f16 (transposed H tile)
  float* dsts = (float*)(sm + 32 * HST * 2);          // 1024 f32
  float* srcs = dsts + 1024;                          // 1024 f32
  float* red  = srcs + 1024;                          // 8 f32
  int bid = blockIdx.x;
  int g = bid / NH, head = bid % NH;
  int tid = threadIdx.x;
  int C = NH * 32;
  // fill transposed tile: 16B global reads, register transpose
  for (int it = tid; it < 4096; it += 256) {
    int j = it & 1023, ci = (it >> 10) << 3;
    v8h hv = *(const v8h*)(H + ((size_t)g * 1024 + j) * C + head * 32 + ci);
#pragma unroll
    for (int u = 0; u < 8; ++u) HsT[(ci + u) * HST + j] = hv[u];
  }
  for (int j = tid; j < 1024; j += 256) {
    dsts[j] = dst[((size_t)g * 1024 + j) * NH + head];
    srcs[j] = src[((size_t)g * 1024 + j) * NH + head];
  }
  __syncthreads();
  // dmax = max_j dst_j (block reduce)
  float dm = -1e30f;
  for (int j = tid; j < 1024; j += 256) dm = fmaxf(dm, dsts[j]);
#pragma unroll
  for (int o = 16; o >= 1; o >>= 1) dm = fmaxf(dm, __shfl_xor(dm, o, 32));
  int wave = tid >> 5, lane = tid & 31;
  if (lane == 0) red[wave] = dm;
  __syncthreads();
  float dmax = red[0];
#pragma unroll
  for (int w = 1; w < 8; ++w) dmax = fmaxf(dmax, red[w]);

  int half = lane >> 4, row = lane & 15, col = lane & 15;
  for (int t = wave; t < 64; t += 8) {               // i-tile of 16 rows
    float si = srcs[t * 16 + row];
    float mi = lrelu(si + dmax);                     // exact row max (monotone lrelu)
    v8f a0 = {}, a1 = {};
    float zp = 0.f;
    for (int jb = 0; jb < 1024; jb += 32) {
      const float* dp = dsts + jb + 16 * half;
      float dv[16];
      *(float4*)(dv + 0)  = *(const float4*)(dp + 0);   // K = 16h   .. 16h+3
      *(float4*)(dv + 4)  = *(const float4*)(dp + 4);   // K = 16h+4 .. 16h+7
      *(float4*)(dv + 8)  = *(const float4*)(dp + 8);   // K = 16h+8 ..
      *(float4*)(dv + 12) = *(const float4*)(dp + 12);
      v16h a;
#pragma unroll
      for (int e = 0; e < 16; ++e) {
        float p = __expf(lrelu(si + dv[e]) - mi);
        zp += p;
        a[e] = (_Float16)p;
      }
      v16h b0 = loadBT16(&HsT[col * HST + jb + 16 * half]);
      v16h b1 = loadBT16(&HsT[(16 + col) * HST + jb + 16 * half]);
      a0 = wmma16(a, b0, a0);
      a1 = wmma16(a, b1, a1);
    }
    float z2 = zp + __shfl_xor(zp, 16, 32);          // full row sum in lanes (row, row+16)
#pragma unroll
    for (int v = 0; v < 8; ++v) {
      int rr = v + 8 * half;
      float Z = __shfl(z2, rr, 32);
      float inv = 1.f / Z;
      float o0 = a0[v] * inv + bias[head * 32 + col];
      float o1 = a1[v] * inv + bias[head * 32 + 16 + col];
      if (RELU) { o0 = fmaxf(o0, 0.f); o1 = fmaxf(o1, 0.f); }
      size_t base = ((size_t)g * 1024 + t * 16 + rr) * C + head * 32;
      if (OUT16) { outh[base + col] = (_Float16)o0; outh[base + 16 + col] = (_Float16)o1; }
      else       { outf[base + col] = o0;           outf[base + 16 + col] = o1; }
    }
  }
}

// ---------------- GAT2 GEMM: h2 = h1relu @ w2  (1024x128x32 per graph) -------
__global__ void k_gat2_gemm(const _Float16* __restrict__ h1r, const _Float16* __restrict__ w2t,
                            float* __restrict__ h2f, _Float16* __restrict__ h2h) {
  int bid = blockIdx.x;
  int g = bid >> 6, tile = bid & 63;
  int lane = threadIdx.x, half = lane >> 4, col = lane & 15, row = lane & 15;
  const _Float16* Ab = h1r + (size_t)g * 1024 * 128 + (tile * 16 + row) * 128 + 16 * half;
  v8f a0 = {}, a1 = {};
  for (int kc = 0; kc < 128; kc += 32) {
    v16h a  = loadA16(Ab + kc);
    v16h b0 = loadBT16(&w2t[col * 128 + kc + 16 * half]);        // w2t: (32,128) = w2^T
    v16h b1 = loadBT16(&w2t[(16 + col) * 128 + kc + 16 * half]);
    a0 = wmma16(a, b0, a0);
    a1 = wmma16(a, b1, a1);
  }
#pragma unroll
  for (int v = 0; v < 8; ++v) {
    int rr = v + 8 * half;
    size_t o = ((size_t)g * 1024 + tile * 16 + rr) * 32;
    h2f[o + col] = a0[v];            h2f[o + 16 + col] = a1[v];
    h2h[o + col] = (_Float16)a0[v];  h2h[o + 16 + col] = (_Float16)a1[v];
  }
}

__global__ void k_srcdst2(const float* __restrict__ h2f, const float* __restrict__ asrc,
                          const float* __restrict__ adst, float* __restrict__ src,
                          float* __restrict__ dst) {
  int idx = blockIdx.x * blockDim.x + threadIdx.x;
  if (idx >= 32 * 1024) return;
  const float* hr = h2f + (size_t)idx * 32;
  float s = 0.f, d = 0.f;
#pragma unroll
  for (int c = 0; c < 32; ++c) { s += hr[c] * asrc[c]; d += hr[c] * adst[c]; }
  src[idx] = s; dst[idx] = d;
}

// ---------------- sf = spatial_out.mean(axis=1) ----------------
__global__ void k_sf_mean(const float* __restrict__ spat, float* __restrict__ sf) {
  int idx = blockIdx.x * blockDim.x + threadIdx.x;   // (b*1024+n)*32+c
  if (idx >= 2 * 1024 * 32) return;
  int b = idx >> 15;
  int n = (idx >> 5) & 1023;
  int c = idx & 31;
  float s = 0.f;
  for (int t = 0; t < 16; ++t)
    s += spat[(((size_t)b * 16 + t) * 1024 + n) * 32 + c];
  sf[idx] = s * (1.f / 16.f);
}

// ---------------- fully fused temporal transformer block (per 16x32 sequence) -
__global__ void k_temporal(const float* __restrict__ spat,
                           const _Float16* __restrict__ wqt, const float* __restrict__ bq,
                           const _Float16* __restrict__ wkt, const float* __restrict__ bk,
                           const _Float16* __restrict__ wvt, const float* __restrict__ bv,
                           const _Float16* __restrict__ wot, const float* __restrict__ bo,
                           const float* __restrict__ ln1g, const float* __restrict__ ln1b,
                           const _Float16* __restrict__ w1t, const float* __restrict__ fb1,
                           const _Float16* __restrict__ w2t, const float* __restrict__ fb2,
                           const float* __restrict__ ln2g, const float* __restrict__ ln2b,
                           float* __restrict__ tf) {
  __shared__ float    Xs[16 * 32];
  __shared__ _Float16 Xh[16 * 32];
  __shared__ _Float16 Qh[16 * 64];
  __shared__ _Float16 Kh[16 * 64];
  __shared__ _Float16 VhT[64 * 32];  // transposed V, cols t=16..31 zero for K=32 WMMA
  __shared__ _Float16 Ph[16 * 32];   // cols 16..31 zero-padded
  __shared__ _Float16 Oh[16 * 64];
  __shared__ _Float16 Yh[16 * 32];
  __shared__ _Float16 Fh[16 * 64];
  const int r = blockIdx.x;
  const int lane = threadIdx.x;
  const int half = lane >> 4, col = lane & 15, row = lane & 15;
  const float* xb = spat + (size_t)r * 512;
  // async copy of the 2KB sequence tile into LDS (ASYNCcnt-tracked)
  {
    unsigned lbase = (unsigned)(uintptr_t)(&Xs[0]);
    unsigned long long gbase = (unsigned long long)(uintptr_t)xb;
#pragma unroll
    for (int t = 0; t < 4; ++t) {
      unsigned loff = lbase + (unsigned)((lane + t * 32) * 16);
      unsigned long long ga = gbase + (unsigned long long)((lane + t * 32) * 16);
      asm volatile("global_load_async_to_lds_b128 %0, %1, off"
                   :: "v"(loff), "v"(ga) : "memory");
    }
    asm volatile("s_wait_asynccnt 0x0" ::: "memory");
  }
  for (int e = lane; e < 2048; e += 32) VhT[e] = (_Float16)0.f;
  __syncthreads();
  for (int e = lane; e < 512; e += 32) Xh[e] = (_Float16)Xs[e];
  __syncthreads();
  v16h ax = loadA16(&Xh[row * 32 + 16 * half]);
  // Q,K,V projections (Q pre-scaled by 1/sqrt(32)); V stored transposed
  for (int w = 0; w < 3; ++w) {
    const _Float16* WT = (w == 0) ? wqt : (w == 1) ? wkt : wvt;  // (64,32) = W^T
    const float* bb    = (w == 0) ? bq  : (w == 1) ? bk  : bv;
    _Float16* dstp     = (w == 0) ? Qh  : Kh;
    const float scale  = (w == 0) ? 0.17677669529663687f : 1.f;
    for (int nt = 0; nt < 4; ++nt) {
      int nc = nt * 16 + col;
      v16h b = loadBT16(&WT[nc * 32 + 16 * half]);
      v8f acc = {};
      acc = wmma16(ax, b, acc);
#pragma unroll
      for (int v = 0; v < 8; ++v) {
        int rr = v + 8 * half;
        _Float16 hv = (_Float16)((acc[v] + bb[nc]) * scale);
        if (w == 2) VhT[nc * 32 + rr] = hv;
        else        dstp[rr * 64 + nc] = hv;
      }
    }
  }
  __syncthreads();
  // per-head attention (S=16 -> single 16x16 logits tile)
  for (int h0 = 0; h0 < 2; ++h0) {
    v16h aq  = loadA16(&Qh[row * 64 + h0 * 32 + 16 * half]);
    v16h bk2 = loadBT16(&Kh[col * 64 + h0 * 32 + 16 * half]);
    v8f lg = {};
    lg = wmma16(aq, bk2, lg);
#pragma unroll
    for (int v = 0; v < 8; ++v) {
      float xv = lg[v];
      float m = xv;
#pragma unroll
      for (int o = 8; o >= 1; o >>= 1) m = fmaxf(m, __shfl_xor(m, o, 16));
      float e = __expf(xv - m);
      float s = e;
#pragma unroll
      for (int o = 8; o >= 1; o >>= 1) s += __shfl_xor(s, o, 16);
      int rr = v + 8 * half;
      Ph[rr * 32 + col] = (_Float16)(e / s);
      Ph[rr * 32 + 16 + col] = (_Float16)0.f;
    }
    __syncthreads();
    v16h ap = loadA16(&Ph[row * 32 + 16 * half]);
    for (int ct = 0; ct < 2; ++ct) {
      v16h bv2 = loadBT16(&VhT[(h0 * 32 + ct * 16 + col) * 32 + 16 * half]);
      v8f oa = {};
      oa = wmma16(ap, bv2, oa);
#pragma unroll
      for (int v = 0; v < 8; ++v)
        Oh[(v + 8 * half) * 64 + h0 * 32 + ct * 16 + col] = (_Float16)oa[v];
    }
    __syncthreads();
  }
  // output projection + residual + LN1
  v16h ao0 = loadA16(&Oh[row * 64 + 16 * half]);
  v16h ao1 = loadA16(&Oh[row * 64 + 32 + 16 * half]);
  float yv[16];
  for (int nt = 0; nt < 2; ++nt) {
    int nc = nt * 16 + col;
    v16h b0 = loadBT16(&wot[nc * 64 + 16 * half]);        // wot: (32,64) = wo^T
    v16h b1 = loadBT16(&wot[nc * 64 + 32 + 16 * half]);
    v8f acc = {};
    acc = wmma16(ao0, b0, acc);
    acc = wmma16(ao1, b1, acc);
#pragma unroll
    for (int v = 0; v < 8; ++v)
      yv[nt * 8 + v] = acc[v] + bo[nc] + Xs[(v + 8 * half) * 32 + nc];
  }
#pragma unroll
  for (int v = 0; v < 8; ++v) {
    float a0 = yv[v], a1 = yv[8 + v];
    float s = a0 + a1;
#pragma unroll
    for (int o = 8; o >= 1; o >>= 1) s += __shfl_xor(s, o, 16);
    float mu = s * (1.f / 32.f);
    float d0 = a0 - mu, d1 = a1 - mu;
    float q = d0 * d0 + d1 * d1;
#pragma unroll
    for (int o = 8; o >= 1; o >>= 1) q += __shfl_xor(q, o, 16);
    float rstd = rsqrtf(q * (1.f / 32.f) + 1e-3f);
    float y0 = d0 * rstd * ln1g[col] + ln1b[col];
    float y1 = d1 * rstd * ln1g[16 + col] + ln1b[16 + col];
    yv[v] = y0; yv[8 + v] = y1;
    int rr = v + 8 * half;
    Yh[rr * 32 + col] = (_Float16)y0;
    Yh[rr * 32 + 16 + col] = (_Float16)y1;
  }
  __syncthreads();
  // FFN up + exact GELU
  v16h ay = loadA16(&Yh[row * 32 + 16 * half]);
  for (int nt = 0; nt < 4; ++nt) {
    int nc = nt * 16 + col;
    v16h b = loadBT16(&w1t[nc * 32 + 16 * half]);         // w1t: (64,32) = ff_w1^T
    v8f acc = {};
    acc = wmma16(ay, b, acc);
#pragma unroll
    for (int v = 0; v < 8; ++v)
      Fh[(v + 8 * half) * 64 + nc] = (_Float16)gelu_exact(acc[v] + fb1[nc]);
  }
  __syncthreads();
  // FFN down + residual + LN2 + fused tf mean over S
  v16h af0 = loadA16(&Fh[row * 64 + 16 * half]);
  v16h af1 = loadA16(&Fh[row * 64 + 32 + 16 * half]);
  float z0[8], z1[8];
  for (int nt = 0; nt < 2; ++nt) {
    int nc = nt * 16 + col;
    v16h bw0 = loadBT16(&w2t[nc * 64 + 16 * half]);       // w2t: (32,64) = ff_w2^T
    v16h bw1 = loadBT16(&w2t[nc * 64 + 32 + 16 * half]);
    v8f acc = {};
    acc = wmma16(af0, bw0, acc);
    acc = wmma16(af1, bw1, acc);
#pragma unroll
    for (int v = 0; v < 8; ++v) {
      float val = acc[v] + fb2[nc] + yv[nt * 8 + v];
      if (nt == 0) z0[v] = val; else z1[v] = val;
    }
  }
  float p0 = 0.f, p1 = 0.f;
#pragma unroll
  for (int v = 0; v < 8; ++v) {
    float a0 = z0[v], a1 = z1[v];
    float s = a0 + a1;
#pragma unroll
    for (int o = 8; o >= 1; o >>= 1) s += __shfl_xor(s, o, 16);
    float mu = s * (1.f / 32.f);
    float d0 = a0 - mu, d1 = a1 - mu;
    float q = d0 * d0 + d1 * d1;
#pragma unroll
    for (int o = 8; o >= 1; o >>= 1) q += __shfl_xor(q, o, 16);
    float rstd = rsqrtf(q * (1.f / 32.f) + 1e-3f);
    p0 += d0 * rstd * ln2g[col] + ln2b[col];
    p1 += d1 * rstd * ln2g[16 + col] + ln2b[16 + col];
  }
  p0 += __shfl_xor(p0, 16, 32);
  p1 += __shfl_xor(p1, 16, 32);
  if (lane < 16) {
    tf[(size_t)r * 32 + lane]      = p0 * (1.f / 16.f);
    tf[(size_t)r * 32 + 16 + lane] = p1 * (1.f / 16.f);
  }
}

// ---------------- cross-attention ----------------
__global__ void k_cross_qkv(const float* __restrict__ sf, const float* __restrict__ tf,
                            const float* __restrict__ wq, const float* __restrict__ bq,
                            const float* __restrict__ wk, const float* __restrict__ bk,
                            const float* __restrict__ wv, const float* __restrict__ bv,
                            float* __restrict__ QF, float* __restrict__ KF,
                            _Float16* __restrict__ Q16, _Float16* __restrict__ K16,
                            _Float16* __restrict__ VT16) {
  int idx = blockIdx.x * blockDim.x + threadIdx.x;   // b*1024+n
  if (idx >= 2048) return;
  int b = idx >> 10, n = idx & 1023;
  const float* s = sf + (size_t)idx * 32;
  const float* t = tf + (size_t)idx * 32;
  for (int hk = 0; hk < 64; ++hk) {
    float q = bq[hk], k = bk[hk], v = bv[hk];
#pragma unroll
    for (int d = 0; d < 32; ++d) {
      q += s[d] * wq[d * 64 + hk];
      k += t[d] * wk[d * 64 + hk];
      v += t[d] * wv[d * 64 + hk];
    }
    q *= 0.17677669529663687f;
    size_t o = (size_t)idx * 64 + hk;
    QF[o] = q; KF[o] = k;
    Q16[o] = (_Float16)q; K16[o] = (_Float16)k;
    VT16[((size_t)b * 64 + hk) * 1024 + n] = (_Float16)v;   // transposed V
  }
}

__global__ void k_cross_pass1(const float* __restrict__ QF, const float* __restrict__ KF,
                              float* __restrict__ mrow, float* __restrict__ zrow) {
  int bid = blockIdx.x;                              // (b*2+h)*1024 + i
  int i = bid & 1023;
  int h = (bid >> 10) & 1;
  int b = bid >> 11;
  int lane = threadIdx.x;
  float qv[32];
  const float* q = QF + ((size_t)(b * 1024 + i)) * 64 + h * 32;
#pragma unroll
  for (int c = 0; c < 32; ++c) qv[c] = q[c];
  const float* Kb = KF + ((size_t)b * 1024) * 64 + h * 32;
  float m = -1e30f;
  for (int t = lane; t < 1024; t += 32) {
    const float* kr = Kb + (size_t)t * 64;
    float d = 0.f;
#pragma unroll
    for (int c = 0; c < 32; ++c) d += qv[c] * kr[c];
    m = fmaxf(m, d);
  }
#pragma unroll
  for (int o = 16; o >= 1; o >>= 1) m = fmaxf(m, __shfl_xor(m, o, 32));
  float z = 0.f;
  for (int t = lane; t < 1024; t += 32) {
    const float* kr = Kb + (size_t)t * 64;
    float d = 0.f;
#pragma unroll
    for (int c = 0; c < 32; ++c) d += qv[c] * kr[c];
    z += __expf(d - m);
  }
#pragma unroll
  for (int o = 16; o >= 1; o >>= 1) z += __shfl_xor(z, o, 32);
  if (lane == 0) { mrow[bid] = m; zrow[bid] = z; }
}

__global__ void k_cross_pass2(const _Float16* __restrict__ Q16, const _Float16* __restrict__ K16,
                              const _Float16* __restrict__ VT16, const float* __restrict__ mrow,
                              const float* __restrict__ zrow, float* __restrict__ OC) {
  __shared__ float m16[16], z16[16];
  __shared__ _Float16 Ph[16 * 32];
  int bid = blockIdx.x;                              // b*128 + h*64 + tile
  int tile = bid & 63;
  int h = (bid >> 6) & 1;
  int b = bid >> 7;
  int lane = threadIdx.x, half = lane >> 4, col = lane & 15, row = lane & 15;
  int i0 = tile * 16;
  if (lane < 16) {
    m16[lane] = mrow[((b * 2 + h) << 10) + i0 + lane];
    z16[lane] = zrow[((b * 2 + h) << 10) + i0 + lane];
  }
  __syncthreads();
  const _Float16* Qb = Q16 + ((size_t)b << 10) * 64 + h * 32;
  const _Float16* Vt = VT16 + ((size_t)b * 64 + h * 32) * 1024;
  v16h aq = loadA16(Qb + (i0 + row) * 64 + 16 * half);
  v8f o0 = {}, o1 = {};
  for (int jb = 0; jb < 1024; jb += 32) {
    for (int nt = 0; nt < 2; ++nt) {
      const _Float16* Kb = K16 + ((size_t)(b * 1024 + jb + nt * 16 + col)) * 64 + h * 32;
      v16h bk = loadBT16(Kb + 16 * half);
      v8f lg = {};
      lg = wmma16(aq, bk, lg);
#pragma unroll
      for (int v = 0; v < 8; ++v) {
        int rr = v + 8 * half;
        Ph[rr * 32 + nt * 16 + col] = (_Float16)__expf(lg[v] - m16[rr]);
      }
    }
    __syncthreads();
    v16h ap = loadA16(&Ph[row * 32 + 16 * half]);
    v16h b0 = loadBT16(Vt + (size_t)col * 1024 + jb + 16 * half);
    v16h b1 = loadBT16(Vt + (size_t)(16 + col) * 1024 + jb + 16 * half);
    o0 = wmma16(ap, b0, o0);
    o1 = wmma16(ap, b1, o1);
    __syncthreads();
  }
#pragma unroll
  for (int v = 0; v < 8; ++v) {
    int rr = v + 8 * half;
    float inv = 1.f / z16[rr];
    size_t base = ((size_t)(b * 1024 + i0 + rr)) * 64 + h * 32;
    OC[base + col] = o0[v] * inv;
    OC[base + 16 + col] = o1[v] * inv;
  }
}

// ---------------- output projection + mean + final dense ----------------
__global__ void k_final(const float* __restrict__ OC, const float* __restrict__ wo,
                        const float* __restrict__ bo, const float* __restrict__ fdw,
                        const float* __restrict__ fdb, float* __restrict__ out) {
  __shared__ float red[256];
  int b = blockIdx.x, tid = threadIdx.x;
  float part = 0.f;
  for (int n = tid; n < 1024; n += 256) {
    const float* o = OC + ((size_t)b * 1024 + n) * 64;
    for (int d = 0; d < 32; ++d) {
      float f = bo[d];
#pragma unroll
      for (int k = 0; k < 64; ++k) f += o[k] * wo[k * 32 + d];
      part += f * fdw[d];
    }
  }
  red[tid] = part;
  __syncthreads();
  for (int s = 128; s > 0; s >>= 1) {
    if (tid < s) red[tid] += red[tid + s];
    __syncthreads();
  }
  if (tid == 0) out[b] = red[0] * (1.f / 1024.f) + fdb[0];
}

extern "C" void kernel_launch(void* const* d_in, const int* in_sizes, int n_in,
                              void* d_out, int out_size, void* d_ws, size_t ws_size,
                              hipStream_t stream) {
  (void)in_sizes; (void)n_in; (void)out_size; (void)ws_size;
  const float* x         = (const float*)d_in[0];
  const float* gat1_w    = (const float*)d_in[1];
  const float* gat1_asrc = (const float*)d_in[2];
  const float* gat1_adst = (const float*)d_in[3];
  const float* gat1_b    = (const float*)d_in[4];
  const float* gat2_w    = (const float*)d_in[5];
  const float* gat2_asrc = (const float*)d_in[6];
  const float* gat2_adst = (const float*)d_in[7];
  const float* gat2_b    = (const float*)d_in[8];
  const float* t_wq = (const float*)d_in[9];
  const float* t_bq = (const float*)d_in[10];
  const float* t_wk = (const float*)d_in[11];
  const float* t_bk = (const float*)d_in[12];
  const float* t_wv = (const float*)d_in[13];
  const float* t_bv = (const float*)d_in[14];
  const float* t_wo = (const float*)d_in[15];
  const float* t_bo = (const float*)d_in[16];
  const float* ln1_g = (const float*)d_in[17];
  const float* ln1_b = (const float*)d_in[18];
  const float* ln2_g = (const float*)d_in[19];
  const float* ln2_b = (const float*)d_in[20];
  const float* ff_w1 = (const float*)d_in[21];
  const float* ff_b1 = (const float*)d_in[22];
  const float* ff_w2 = (const float*)d_in[23];
  const float* ff_b2 = (const float*)d_in[24];
  const float* c_wq = (const float*)d_in[25];
  const float* c_bq = (const float*)d_in[26];
  const float* c_wk = (const float*)d_in[27];
  const float* c_bk = (const float*)d_in[28];
  const float* c_wv = (const float*)d_in[29];
  const float* c_bv = (const float*)d_in[30];
  const float* c_wo = (const float*)d_in[31];
  const float* c_bo = (const float*)d_in[32];
  const float* fd_w = (const float*)d_in[33];
  const float* fd_b = (const float*)d_in[34];

  char* ws = (char*)d_ws;
  size_t off = 0;
  auto alloc = [&](size_t bytes) {
    char* p = ws + off;
    off = (off + bytes + 255) & ~(size_t)255;
    return p;
  };
  _Float16* H1h  = (_Float16*)alloc(32ull * 1024 * 128 * 2);
  float*    SRC1 = (float*)alloc(32ull * 1024 * 4 * 4);
  float*    DST1 = (float*)alloc(32ull * 1024 * 4 * 4);
  _Float16* H1R  = (_Float16*)alloc(32ull * 1024 * 128 * 2);
  float*    H2F  = (float*)alloc(32ull * 1024 * 32 * 4);
  _Float16* H2H  = (_Float16*)alloc(32ull * 1024 * 32 * 2);
  float*    SRC2 = (float*)alloc(32ull * 1024 * 4);
  float*    DST2 = (float*)alloc(32ull * 1024 * 4);
  float*    SPAT = (float*)alloc(2ull * 16 * 1024 * 32 * 4);
  float*    TF   = (float*)alloc(2ull * 1024 * 32 * 4);
  float*    SF   = (float*)alloc(2ull * 1024 * 32 * 4);
  float*    QF   = (float*)alloc(2ull * 1024 * 64 * 4);
  float*    KF   = (float*)alloc(2ull * 1024 * 64 * 4);
  _Float16* Q16  = (_Float16*)alloc(2ull * 1024 * 64 * 2);
  _Float16* K16  = (_Float16*)alloc(2ull * 1024 * 64 * 2);
  _Float16* VT16 = (_Float16*)alloc(2ull * 64 * 1024 * 2);
  float*    MROW = (float*)alloc(2ull * 2 * 1024 * 4);
  float*    ZROW = (float*)alloc(2ull * 2 * 1024 * 4);
  float*    OCb  = (float*)alloc(2ull * 1024 * 64 * 4);
  _Float16* W2T  = (_Float16*)alloc(4096 * 2);
  _Float16* TWQT = (_Float16*)alloc(2048 * 2);
  _Float16* TWKT = (_Float16*)alloc(2048 * 2);
  _Float16* TWVT = (_Float16*)alloc(2048 * 2);
  _Float16* TWOT = (_Float16*)alloc(2048 * 2);
  _Float16* FW1T = (_Float16*)alloc(2048 * 2);
  _Float16* FW2T = (_Float16*)alloc(2048 * 2);

  const size_t AGG_LDS = 32 * HST * 2 + 1024 * 4 + 1024 * 4 + 64;
  (void)hipFuncSetAttribute((const void*)&k_gat_agg<true, true>,
                            hipFuncAttributeMaxDynamicSharedMemorySize, (int)AGG_LDS);
  (void)hipFuncSetAttribute((const void*)&k_gat_agg<false, false>,
                            hipFuncAttributeMaxDynamicSharedMemorySize, (int)AGG_LDS);

  // transposed f16 weight copies (contiguous B fragments)
  k_f32_to_f16_t<<<16, 256, 0, stream>>>(gat2_w, W2T, 128, 32);
  k_f32_to_f16_t<<<8, 256, 0, stream>>>(t_wq, TWQT, 32, 64);
  k_f32_to_f16_t<<<8, 256, 0, stream>>>(t_wk, TWKT, 32, 64);
  k_f32_to_f16_t<<<8, 256, 0, stream>>>(t_wv, TWVT, 32, 64);
  k_f32_to_f16_t<<<8, 256, 0, stream>>>(t_wo, TWOT, 64, 32);
  k_f32_to_f16_t<<<8, 256, 0, stream>>>(ff_w1, FW1T, 32, 64);
  k_f32_to_f16_t<<<8, 256, 0, stream>>>(ff_w2, FW2T, 64, 32);

  k_gat1_prep<<<128, 256, 0, stream>>>(x, gat1_w, gat1_asrc, gat1_adst, H1h, SRC1, DST1);
  k_gat_agg<true, true><<<128, 256, AGG_LDS, stream>>>(H1h, SRC1, DST1, gat1_b,
                                                       nullptr, H1R, 4);
  k_gat2_gemm<<<2048, 32, 0, stream>>>(H1R, W2T, H2F, H2H);
  k_srcdst2<<<128, 256, 0, stream>>>(H2F, gat2_asrc, gat2_adst, SRC2, DST2);
  k_gat_agg<false, false><<<32, 256, AGG_LDS, stream>>>(H2H, SRC2, DST2, gat2_b,
                                                        SPAT, nullptr, 1);
  k_sf_mean<<<256, 256, 0, stream>>>(SPAT, SF);
  k_temporal<<<2048, 32, 0, stream>>>(SPAT, TWQT, t_bq, TWKT, t_bk, TWVT, t_bv, TWOT, t_bo,
                                      ln1_g, ln1_b, FW1T, ff_b1, FW2T, ff_b2, ln2_g, ln2_b, TF);
  k_cross_qkv<<<8, 256, 0, stream>>>(SF, TF, c_wq, c_bq, c_wk, c_bk, c_wv, c_bv,
                                     QF, KF, Q16, K16, VT16);
  k_cross_pass1<<<4096, 32, 0, stream>>>(QF, KF, MROW, ZROW);
  k_cross_pass2<<<256, 32, 0, stream>>>(Q16, K16, VT16, MROW, ZROW, OCb);
  k_final<<<2, 256, 0, stream>>>(OCb, c_wo, c_bo, fd_w, fd_b, (float*)d_out);
}
